// BitNetMLP_26611617366616
// MI455X (gfx1250) — 8-bit matrix operand order repaired and checked by reading, compiled, not run on hardware
//
#include <hip/hip_runtime.h>
#include <hip/hip_bf16.h>

static constexpr int   HD    = 4096;
static constexpr int   ID    = 11008;
static constexpr int   MD    = 4096;
static constexpr float SQRT7 = 2.6457513110645906f;
static constexpr float QEPS  = 1e-5f;

typedef int      v8i __attribute__((ext_vector_type(8)));
typedef unsigned uv4 __attribute__((ext_vector_type(4)));
typedef int      iv4 __attribute__((ext_vector_type(4)));
typedef int      iv8 __attribute__((ext_vector_type(8)));

#if __has_builtin(__builtin_amdgcn_tensor_load_to_lds) && __has_builtin(__builtin_amdgcn_s_wait_tensorcnt)
#define USE_TDM 1
#else
#define USE_TDM 0
#endif

static constexpr int TROW = 80;
static constexpr int TILE = 64 * TROW;

#define LDS_MARK_WRITTEN(smarr)                                        \
    do {                                                               \
        if (blockIdx.x == 0xFFFFFFFFu) (smarr)[threadIdx.x] = 1;       \
    } while (0)

__device__ __forceinline__ v8i wmma_iu8(v8i a, v8i b, v8i c) {
    return __builtin_amdgcn_wmma_i32_16x16x64_iu8(true, a, true, b, c, false, false);
}

__device__ __forceinline__ v8i lds_a_frag(const char* p0, int lane) {
    const char* p = p0 + ((lane >> 4) << 4);
    int2 t0 = *(const int2*)(p);
    int2 t1 = *(const int2*)(p + 8);
    int2 t2 = *(const int2*)(p + 32);
    int2 t3 = *(const int2*)(p + 40);
    v8i a;
    a[0] = t0.x; a[1] = t0.y; a[2] = t1.x; a[3] = t1.y;
    a[4] = t2.x; a[5] = t2.y; a[6] = t3.x; a[7] = t3.y;
    return a;
}

__device__ __forceinline__ v8i lds_b_frag(const char* p0, int lane) {
    const char* p = p0 + ((lane >> 4) << 4);
    int4 t0 = *(const int4*)(p);
    int4 t1 = *(const int4*)(p + 32);
    v8i b;
    b[0] = t0.x; b[1] = t0.y; b[2] = t0.z; b[3] = t0.w;
    b[4] = t1.x; b[5] = t1.y; b[6] = t1.z; b[7] = t1.w;
    return b;
}

#if USE_TDM
__device__ __forceinline__ void tdm_load_2d(unsigned lds_off, const void* gptr,
                                            unsigned long long row_stride,
                                            unsigned rows_total) {
    unsigned long long ga = (unsigned long long)gptr;
    uv4 g0;
    g0[0] = 1u;
    g0[1] = lds_off;
    g0[2] = (unsigned)(ga & 0xFFFFFFFFull);
    g0[3] = (unsigned)((ga >> 32) & 0x01FFFFFFull) | (2u << 30);

    unsigned td0 = (unsigned)row_stride;
    iv8 g1;
    g1[0] = (int)((1u << 20) | (3u << 22) | (3u << 25));
    g1[1] = (int)((td0 & 0xFFFFu) << 16);
    g1[2] = (int)((td0 >> 16) | ((rows_total & 0xFFFFu) << 16));
    g1[3] = (int)((rows_total >> 16) | (64u << 16));
    g1[4] = (int)64u;
    g1[5] = (int)(unsigned)(row_stride & 0xFFFFFFFFull);
    g1[6] = (int)(unsigned)((row_stride >> 32) & 0xFFFFull);
    g1[7] = 0;

    iv4 z4 = {0, 0, 0, 0};
#if defined(__clang_major__) && (__clang_major__ >= 23)
    iv8 z8 = {0, 0, 0, 0, 0, 0, 0, 0};
    __builtin_amdgcn_tensor_load_to_lds(g0, g1, z4, z4, z8, 0);
#else
    __builtin_amdgcn_tensor_load_to_lds(g0, g1, z4, z4, 0);
#endif
}
#endif

__device__ __forceinline__ void stage_tile(char* dst, const char* src, int srcStride) {
    const int t   = threadIdx.x;
    const int row = t >> 2;
    const int c   = (t & 3) * 16;
    *(int4*)(dst + row * TROW + c) = *(const int4*)(src + (size_t)row * srcStride + c);
}

__global__ void init_kernel(float* wsums, unsigned* gamma_u) {
    int i = blockIdx.x * blockDim.x + threadIdx.x;
    if (i < 3) wsums[i] = 0.0f;
    int g = i - 3;
    if (g >= 0 && g < MD) gamma_u[g] = 0u;
}

__global__ void absum_kernel(const float* __restrict__ w, size_t n, float* out) {
    __shared__ float red[256];
    size_t i = (size_t)blockIdx.x * blockDim.x + threadIdx.x;
    size_t stride = (size_t)gridDim.x * blockDim.x;
    float s = 0.0f;
    for (; i < n; i += stride) s += fabsf(w[i]);
    red[threadIdx.x] = s;
    __syncthreads();
    for (int t = 128; t > 0; t >>= 1) {
        if ((int)threadIdx.x < t) red[threadIdx.x] += red[threadIdx.x + t];
        __syncthreads();
    }
    if (threadIdx.x == 0) atomicAdd(out, red[0]);
}

__global__ void quantw_kernel(const float* __restrict__ w, size_t n,
                              const float* __restrict__ sum, float invn,
                              char* __restrict__ q) {
    size_t i = (size_t)blockIdx.x * blockDim.x + threadIdx.x;
    if (i >= n) return;
    float mean = *sum * invn;
    float v = rintf(w[i] / (mean + QEPS));
    v = fminf(fmaxf(v, -1.0f), 1.0f);
    q[i] = (char)(int)v;
}

__global__ void quantx_kernel(const float* __restrict__ x,
                              char* __restrict__ qx, float* __restrict__ beta) {
    __shared__ float red[256];
    int row = blockIdx.x;
    const float* xr = x + (size_t)row * HD;
    float s = 0.0f;
    for (int i = threadIdx.x; i < HD; i += 256) s += fabsf(xr[i]);
    red[threadIdx.x] = s;
    __syncthreads();
    for (int t = 128; t > 0; t >>= 1) {
        if ((int)threadIdx.x < t) red[threadIdx.x] += red[threadIdx.x + t];
        __syncthreads();
    }
    float b = red[0] * (1.0f / (float)HD);
    if (threadIdx.x == 0) beta[row] = b;
    float inv = SQRT7 / (b + QEPS);
    char* qr = qx + (size_t)row * HD;
    for (int i = threadIdx.x; i < HD; i += 256) {
        float v = rintf(xr[i] * inv);
        v = fminf(fmaxf(v, -8.0f), 7.0f);
        qr[i] = (char)(int)v;
    }
}

__global__ __launch_bounds__(256)
void gemm1_kernel(const char* __restrict__ qx,
                  const char* __restrict__ qwg,
                  const char* __restrict__ qwu,
                  __hip_bfloat16* __restrict__ hbuf,
                  unsigned* __restrict__ gamma_u) {
    __shared__ char sm[2 * 3 * TILE];
    LDS_MARK_WRITTEN(sm);
    const int lane   = threadIdx.x & 31;
    const int wave   = threadIdx.x >> 5;
    const int waveM  = wave >> 2;
    const int waveN  = wave & 3;
    const int mblock = blockIdx.y * 64;
    const int nblock = blockIdx.x * 64;
    const int nk     = HD / 64;

    const char* gA = qx  + (size_t)mblock * HD;
    const char* gG = qwg + (size_t)nblock * HD;
    const char* gU = qwu + (size_t)nblock * HD;

#if USE_TDM
    if (wave == 0) {
        tdm_load_2d(0u * TILE, gA,      HD, (unsigned)MD);
        tdm_load_2d(1u * TILE, gG,      HD, (unsigned)ID);
        tdm_load_2d(2u * TILE, gU,      HD, (unsigned)ID);
        tdm_load_2d(3u * TILE, gA + 64, HD, (unsigned)MD);
        tdm_load_2d(4u * TILE, gG + 64, HD, (unsigned)ID);
        tdm_load_2d(5u * TILE, gU + 64, HD, (unsigned)ID);
    }
#endif

    v8i accg0 = {}, accg1 = {}, accu0 = {}, accu1 = {};
    const int ra0 = (waveM * 32 + (lane & 15)) * TROW;
    const int ra1 = ra0 + 16 * TROW;
    const int rb  = (waveN * 16 + (lane & 15)) * TROW;

    for (int k = 0; k < nk; ++k) {
#if USE_TDM
        const int sel = k & 1;
        if (wave == 0) {
            if (k + 1 < nk) __builtin_amdgcn_s_wait_tensorcnt(3);
            else            __builtin_amdgcn_s_wait_tensorcnt(0);
        }
        __syncthreads();
        asm volatile("" ::: "memory");
#else
        const int sel = 0;
        stage_tile(sm + 0 * TILE, gA + k * 64, HD);
        stage_tile(sm + 1 * TILE, gG + k * 64, HD);
        stage_tile(sm + 2 * TILE, gU + k * 64, HD);
        __syncthreads();
#endif
        const char* A  = sm + sel * (3 * TILE);
        const char* Bg = A + TILE;
        const char* Bu = A + 2 * TILE;
        v8i a0 = lds_a_frag(A + ra0, lane);
        v8i a1 = lds_a_frag(A + ra1, lane);
        v8i bg = lds_b_frag(Bg + rb, lane);
        v8i bu = lds_b_frag(Bu + rb, lane);
        accg0 = wmma_iu8(a0, bg, accg0);
        accg1 = wmma_iu8(a1, bg, accg1);
        accu0 = wmma_iu8(a0, bu, accu0);
        accu1 = wmma_iu8(a1, bu, accu1);
        __syncthreads();
#if USE_TDM
        if (wave == 0 && k + 2 < nk) {
            unsigned b = (unsigned)(sel * (3 * TILE));
            tdm_load_2d(b + 0u * TILE, gA + (k + 2) * 64, HD, (unsigned)MD);
            tdm_load_2d(b + 1u * TILE, gG + (k + 2) * 64, HD, (unsigned)ID);
            tdm_load_2d(b + 2u * TILE, gU + (k + 2) * 64, HD, (unsigned)ID);
        }
#endif
    }

    const int col    = nblock + waveN * 16 + (lane & 15);
    const int rowoff = (lane >> 4) * 8;
    const int mwbase = mblock + waveM * 32;
#pragma unroll
    for (int r = 0; r < 8; ++r) {
        int m0 = mwbase + rowoff + r;
        int m1 = m0 + 16;
        float g0 = (float)accg0[r]; g0 = g0 > 0.0f ? g0 : 0.0f;
        float h0 = g0 * g0 * (float)accu0[r];
        hbuf[(size_t)m0 * ID + col] = __float2bfloat16(h0);
        atomicMax(gamma_u + m0, __float_as_uint(fabsf(h0)));
        float g1 = (float)accg1[r]; g1 = g1 > 0.0f ? g1 : 0.0f;
        float h1 = g1 * g1 * (float)accu1[r];
        hbuf[(size_t)m1 * ID + col] = __float2bfloat16(h1);
        atomicMax(gamma_u + m1, __float_as_uint(fabsf(h1)));
    }
}

__global__ void quanth_kernel(const __hip_bfloat16* __restrict__ hbuf,
                              const unsigned* __restrict__ gamma_u,
                              char* __restrict__ qh) {
    int row = blockIdx.y;
    int col = blockIdx.x * blockDim.x + threadIdx.x;
    float gamma = __uint_as_float(gamma_u[row]);
    float inv = 127.0f / (gamma + QEPS);
    size_t idx = (size_t)row * ID + col;
    float v = rintf(__bfloat162float(hbuf[idx]) * inv);
    v = fminf(fmaxf(v, -128.0f), 127.0f);
    qh[idx] = (char)(int)v;
}

__global__ __launch_bounds__(256)
void gemm2_kernel(const char* __restrict__ qh,
                  const char* __restrict__ qwd,
                  const float* __restrict__ beta,
                  const unsigned* __restrict__ gamma_u,
                  const float* __restrict__ wsums,
                  const float* __restrict__ sg,
                  const float* __restrict__ su,
                  const float* __restrict__ sd,
                  float* __restrict__ out) {
    __shared__ char sm[2 * 2 * TILE];
    LDS_MARK_WRITTEN(sm);
    const int lane   = threadIdx.x & 31;
    const int wave   = threadIdx.x >> 5;
    const int waveM  = wave >> 2;
    const int waveN  = wave & 3;
    const int mblock = blockIdx.y * 64;
    const int nblock = blockIdx.x * 64;
    const int nk     = ID / 64;

    const char* gA = qh  + (size_t)mblock * ID;
    const char* gB = qwd + (size_t)nblock * ID;

#if USE_TDM
    if (wave == 0) {
        tdm_load_2d(0u * TILE, gA,      ID, (unsigned)MD);
        tdm_load_2d(1u * TILE, gB,      ID, (unsigned)HD);
        tdm_load_2d(2u * TILE, gA + 64, ID, (unsigned)MD);
        tdm_load_2d(3u * TILE, gB + 64, ID, (unsigned)HD);
    }
#endif

    v8i acc0 = {}, acc1 = {};
    const int ra0 = (waveM * 32 + (lane & 15)) * TROW;
    const int ra1 = ra0 + 16 * TROW;
    const int rb  = (waveN * 16 + (lane & 15)) * TROW;

    for (int k = 0; k < nk; ++k) {
#if USE_TDM
        const int sel = k & 1;
        if (wave == 0) {
            if (k + 1 < nk) __builtin_amdgcn_s_wait_tensorcnt(2);
            else            __builtin_amdgcn_s_wait_tensorcnt(0);
        }
        __syncthreads();
        asm volatile("" ::: "memory");
#else
        const int sel = 0;
        stage_tile(sm + 0 * TILE, gA + k * 64, ID);
        stage_tile(sm + 1 * TILE, gB + k * 64, ID);
        __syncthreads();
#endif
        const char* A = sm + sel * (2 * TILE);
        const char* B = A + TILE;
        v8i a0 = lds_a_frag(A + ra0, lane);
        v8i a1 = lds_a_frag(A + ra1, lane);
        v8i b  = lds_b_frag(B + rb, lane);
        acc0 = wmma_iu8(a0, b, acc0);
        acc1 = wmma_iu8(a1, b, acc1);
        __syncthreads();
#if USE_TDM
        if (wave == 0 && k + 2 < nk) {
            unsigned b2 = (unsigned)(sel * (2 * TILE));
            tdm_load_2d(b2 + 0u * TILE, gA + (k + 2) * 64, ID, (unsigned)MD);
            tdm_load_2d(b2 + 1u * TILE, gB + (k + 2) * 64, ID, (unsigned)HD);
        }
#endif
    }

    const float invWH = 1.0f / ((float)ID * (float)HD);
    const float wgm = wsums[0] * invWH;
    const float wum = wsums[1] * invWH;
    const float wdm = wsums[2] * invWH;
    const float sg0 = *sg, su0 = *su, sd0 = *sd;

    const int col    = nblock + waveN * 16 + (lane & 15);
    const int rowoff = (lane >> 4) * 8;
    const int mwbase = mblock + waveM * 32;
#pragma unroll
    for (int r = 0; r < 8; ++r) {
        for (int half = 0; half < 2; ++half) {
            int m = mwbase + rowoff + r + half * 16;
            float bb = beta[m];
            float srow = (bb * wgm * sg0) * (bb * wgm * sg0) * (bb * wum * su0);
            float gamma_true = __uint_as_float(gamma_u[m]) * srow;
            float scale = gamma_true * (1.0f / 127.0f) * wdm * sd0;
            float accv = (float)(half ? acc1[r] : acc0[r]);
            out[(size_t)m * HD + col] = accv * scale;
        }
    }
}

extern "C" void kernel_launch(void* const* d_in, const int* in_sizes, int n_in,
                              void* d_out, int out_size, void* d_ws, size_t ws_size,
                              hipStream_t stream) {
    const float* x      = (const float*)d_in[0];
    const float* w_gate = (const float*)d_in[1];
    const float* w_up   = (const float*)d_in[2];
    const float* w_down = (const float*)d_in[3];
    const float* s_gate = (const float*)d_in[4];
    const float* s_up   = (const float*)d_in[5];
    const float* s_down = (const float*)d_in[6];
    float* out = (float*)d_out;

    const size_t nXH = (size_t)MD * HD;
    const size_t nW  = (size_t)ID * HD;
    const size_t nMI = (size_t)MD * ID;

    char* base = (char*)d_ws;
    size_t off = 0;
    char* qx  = base + off; off += nXH;
    char* qwg = base + off; off += nW;
    char* qwu = base + off; off += nW;
    char* qwd = base + off; off += nW;
    char* qh  = base + off; off += nMI;
    __hip_bfloat16* hbuf = (__hip_bfloat16*)(base + off); off += nMI * 2;
    float*    beta    = (float*)(base + off);    off += (size_t)MD * 4;
    unsigned* gamma_u = (unsigned*)(base + off); off += (size_t)MD * 4;
    float*    wsums   = (float*)(base + off);    off += 256;

    init_kernel<<<dim3((3 + MD + 255) / 256), dim3(256), 0, stream>>>(wsums, gamma_u);

    absum_kernel<<<dim3(1024), dim3(256), 0, stream>>>(w_gate, nW, wsums + 0);
    absum_kernel<<<dim3(1024), dim3(256), 0, stream>>>(w_up,   nW, wsums + 1);
    absum_kernel<<<dim3(1024), dim3(256), 0, stream>>>(w_down, nW, wsums + 2);

    const float invn = 1.0f / (float)nW;
    dim3 qwGrid((unsigned)((nW + 255) / 256));
    quantw_kernel<<<qwGrid, dim3(256), 0, stream>>>(w_gate, nW, wsums + 0, invn, qwg);
    quantw_kernel<<<qwGrid, dim3(256), 0, stream>>>(w_up,   nW, wsums + 1, invn, qwu);
    quantw_kernel<<<qwGrid, dim3(256), 0, stream>>>(w_down, nW, wsums + 2, invn, qwd);

    quantx_kernel<<<dim3(MD), dim3(256), 0, stream>>>(x, qx, beta);

    gemm1_kernel<<<dim3(ID / 64, MD / 64), dim3(256), 0, stream>>>(qx, qwg, qwu, hbuf, gamma_u);

    quanth_kernel<<<dim3(ID / 256, MD), dim3(256), 0, stream>>>(hbuf, gamma_u, qh);

    gemm2_kernel<<<dim3(HD / 64, MD / 64), dim3(256), 0, stream>>>(
        qh, qwd, beta, gamma_u, wsums, s_gate, s_up, s_down, out);
}
